// LstmModel_8650064134261
// MI455X (gfx1250) — matrix operand order repaired and checked by reading, not compiled, not run on hardware
//
#include <hip/hip_runtime.h>

typedef __attribute__((ext_vector_type(16))) _Float16 v16h;
typedef __attribute__((ext_vector_type(8)))  float    v8f;

#define T_LEN 512
#define BATCH 1024
#define FDIM  16
#define HDIM  32

__device__ __forceinline__ float fast_rcp(float x) {
    return __builtin_amdgcn_rcpf(x);
}
__device__ __forceinline__ float fast_sig(float x) {
    return fast_rcp(1.0f + __expf(-x));
}
__device__ __forceinline__ float fast_tanh(float x) {
    float e = __expf(2.0f * x);
    return 1.0f - 2.0f * fast_rcp(e + 1.0f);
}

// Block = 2 waves = one (16-batch-row tile x direction).
// Gate tiles factor as {gate q: i,f,g,o} x {h-col group p: 0,1}; the cell
// update for col group p needs only tiles {p, 2+p, 4+p, 6+p}. Wave w owns
// col group w: 8 WMMAs/step, private c-state, writes its 16-col half of the
// shared LDS h-tile. Only h is shared (via LDS) -> 2 barriers/step.
__launch_bounds__(64)
__global__ void lstm_scan_kernel(const float* __restrict__ x,
                                 const float* __restrict__ W_ih_f,
                                 const float* __restrict__ W_hh_f,
                                 const float* __restrict__ b_f,
                                 const float* __restrict__ W_ih_b,
                                 const float* __restrict__ W_hh_b,
                                 const float* __restrict__ b_b,
                                 float* __restrict__ hidden)   // (B, 2H) f32
{
    const int tile = blockIdx.x;           // 0..63 : batch tile of 16 rows
    const int dir  = blockIdx.y;           // 0 fwd, 1 bwd
    const float* W_ih = dir ? W_ih_b : W_ih_f;
    const float* W_hh = dir ? W_hh_b : W_hh_f;
    const float* bias = dir ? b_b   : b_f;

    const int w    = threadIdx.x >> 5;     // wave id = h-col group (0/1)
    const int lane = threadIdx.x & 31;
    const int half = lane >> 4;            // 0: lanes 0-15, 1: lanes 16-31
    const int lm   = lane & 15;

    // ---- Pack resident B-operands (32x16 f16, K x N) -------------------
    // B layout: elem e<8 -> K = half*8+e ; elem e>=8 -> K = 16+half*8+(e-8);
    // N = lane%16.  Wave w's gate-q tile is global tile j = 2q + w.
    v16h Bhh[4], Bih[4];
    v8f  biasC[4];                         // persistent bias splat = WMMA C in
    #pragma unroll
    for (int q = 0; q < 4; ++q) {
        const int j = 2 * q + w;
        const float* wh = W_hh + (size_t)(j * 16 + lm) * HDIM;
        const float* wi = W_ih + (size_t)(j * 16 + lm) * FDIM;
        #pragma unroll
        for (int e = 0; e < 16; ++e) {
            Bhh[q][e] = (_Float16)wh[(e < 8) ? (half * 8 + e) : (16 + half * 8 + (e - 8))];
            // x-projection: K=0..15 real, K=16..31 zero-padded
            Bih[q][e] = (e < 8) ? (_Float16)wi[half * 8 + e] : (_Float16)0.0f;
        }
        const float bv = bias[j * 16 + lm];
        #pragma unroll
        for (int r = 0; r < 8; ++r) biasC[q][r] = bv;
    }

    // Shared h tile, row-major 16x32 f16; doubles as the C->A layout
    // transpose between steps. Wave w writes cols [w*16, w*16+16).
    __shared__ _Float16 sh_h[16 * HDIM];
    for (int idx = threadIdx.x; idx < 16 * HDIM; idx += 64)
        sh_h[idx] = (_Float16)0.0f;
    __syncthreads();

    // c state for this wave's 16 h-columns, C-matrix layout (8 rows/lane)
    float cst[8];
    #pragma unroll
    for (int r = 0; r < 8; ++r) cst[r] = 0.0f;

    const int row = tile * 16 + lm;
    const float* xbase = x + (size_t)row * FDIM + half * 8;
    const size_t xstep = (size_t)BATCH * FDIM;

    // Software pipeline: preload x for t=0.
    int tt = dir ? (T_LEN - 1) : 0;
    float4 xp0 = reinterpret_cast<const float4*>(xbase + (size_t)tt * xstep)[0];
    float4 xp1 = reinterpret_cast<const float4*>(xbase + (size_t)tt * xstep)[1];

    for (int t = 0; t < T_LEN; ++t) {
        // ---- A operand: x tile (16x32 f16, K=feature zero-padded) -----
        // A layout: elem p<8 -> K = half*8+p ; elem p>=8 -> K = 16+half*8+(p-8)
        v16h xA;
        xA[0] = (_Float16)xp0.x; xA[1] = (_Float16)xp0.y;
        xA[2] = (_Float16)xp0.z; xA[3] = (_Float16)xp0.w;
        xA[4] = (_Float16)xp1.x; xA[5] = (_Float16)xp1.y;
        xA[6] = (_Float16)xp1.z; xA[7] = (_Float16)xp1.w;
        #pragma unroll
        for (int p = 8; p < 16; ++p) xA[p] = (_Float16)0.0f;

        // Issue next step's x loads now; latency hides under WMMA/VALU.
        {
            int tn  = (t + 1 < T_LEN) ? t + 1 : t;
            int ttn = dir ? (T_LEN - 1 - tn) : tn;
            const float* xpn = xbase + (size_t)ttn * xstep;
            xp0 = reinterpret_cast<const float4*>(xpn)[0];
            xp1 = reinterpret_cast<const float4*>(xpn)[1];
        }

        // ---- A operand: h tile from shared LDS ------------------------
        v16h hA;
        #pragma unroll
        for (int p = 0; p < 8; ++p) hA[p]     = sh_h[lm * HDIM + half * 8 + p];
        #pragma unroll
        for (int p = 0; p < 8; ++p) hA[8 + p] = sh_h[lm * HDIM + 16 + half * 8 + p];
        __syncthreads();   // all reads done before this step's h writes

        // ---- gates = bias + x*W_ih^T + h*W_hh^T  (8 WMMAs/wave) -------
        v8f acc[4];
        #pragma unroll
        for (int q = 0; q < 4; ++q) {
            v8f a = __builtin_amdgcn_wmma_f32_16x16x32_f16(
                        false, xA, false, Bih[q], (short)0, biasC[q], false, false);
            acc[q] = __builtin_amdgcn_wmma_f32_16x16x32_f16(
                        false, hA, false, Bhh[q], (short)0, a, false, false);
        }

        // ---- fused LSTM cell for this wave's 16 h-columns -------------
        #pragma unroll
        for (int r = 0; r < 8; ++r) {
            float iv = fast_sig (acc[0][r]);
            float fv = fast_sig (acc[1][r]);
            float gv = fast_tanh(acc[2][r]);
            float ov = fast_sig (acc[3][r]);
            float cv = fv * cst[r] + iv * gv;
            cst[r] = cv;
            float hn = ov * fast_tanh(cv);
            // C layout: row M = r + 8*half, col = w*16 + lane%16
            sh_h[(r + 8 * half) * HDIM + w * 16 + lm] = (_Float16)hn;
        }
        __syncthreads();   // h writes visible before next step's reads
    }

    // ---- emit final hidden: fwd -> cols 0..31, bwd -> cols 32..63 -----
    for (int idx = threadIdx.x; idx < 16 * HDIM; idx += 64) {
        int m = idx >> 5, k = idx & 31;
        hidden[(size_t)(tile * 16 + m) * (2 * HDIM) + dir * HDIM + k] =
            (float)sh_h[idx];
    }
}

// FC head + softmax: (1024,64)@(64,8)+b -> flatten(8192) -> (8,8192)@flat+b.
// Tiny (0.5 MFLOP, 256 KB traffic): one 1024-thread block, deterministic
// shuffle + LDS tree reduction (no float atomics).
__launch_bounds__(1024)
__global__ void head_kernel(const float* __restrict__ hidden,
                            const float* __restrict__ W_fc_h,
                            const float* __restrict__ b_fc_h,
                            const float* __restrict__ W_fc_o,
                            const float* __restrict__ b_fc_o,
                            float* __restrict__ out)
{
    const int tid = threadIdx.x;   // == batch row b

    float fc[8];
    #pragma unroll
    for (int o = 0; o < 8; ++o) fc[o] = b_fc_h[o];
    const float* hp = hidden + (size_t)tid * 64;
    for (int k = 0; k < 64; ++k) {
        float hv = hp[k];
        #pragma unroll
        for (int o = 0; o < 8; ++o) fc[o] += hv * W_fc_h[o * 64 + k];
    }

    float part[8];
    #pragma unroll
    for (int o = 0; o < 8; ++o) {
        float s = 0.0f;
        #pragma unroll
        for (int j = 0; j < 8; ++j)
            s += W_fc_o[(size_t)o * (BATCH * 8) + tid * 8 + j] * fc[j];
        part[o] = s;
    }

    #pragma unroll
    for (int off = 16; off > 0; off >>= 1) {
        #pragma unroll
        for (int o = 0; o < 8; ++o)
            part[o] += __shfl_down(part[o], off, 32);
    }
    __shared__ float wred[32][8];
    if ((tid & 31) == 0) {
        #pragma unroll
        for (int o = 0; o < 8; ++o) wred[tid >> 5][o] = part[o];
    }
    __syncthreads();

    __shared__ float logits[8];
    if (tid < 8) {
        float s = b_fc_o[tid];
        for (int wv = 0; wv < 32; ++wv) s += wred[wv][tid];
        logits[tid] = s;
    }
    __syncthreads();

    if (tid == 0) {
        float m = logits[0];
        #pragma unroll
        for (int o = 1; o < 8; ++o) m = fmaxf(m, logits[o]);
        float e[8], sum = 0.0f;
        #pragma unroll
        for (int o = 0; o < 8; ++o) { e[o] = __expf(logits[o] - m); sum += e[o]; }
        float inv = fast_rcp(sum);
        #pragma unroll
        for (int o = 0; o < 8; ++o) out[o] = e[o] * inv;
    }
}

extern "C" void kernel_launch(void* const* d_in, const int* in_sizes, int n_in,
                              void* d_out, int out_size, void* d_ws, size_t ws_size,
                              hipStream_t stream) {
    (void)in_sizes; (void)n_in; (void)out_size; (void)ws_size;
    const float* x      = (const float*)d_in[0];
    const float* W_ih_f = (const float*)d_in[1];
    const float* W_hh_f = (const float*)d_in[2];
    const float* b_f    = (const float*)d_in[3];
    const float* W_ih_b = (const float*)d_in[4];
    const float* W_hh_b = (const float*)d_in[5];
    const float* b_b    = (const float*)d_in[6];
    const float* W_fc_h = (const float*)d_in[7];
    const float* b_fc_h = (const float*)d_in[8];
    const float* W_fc_o = (const float*)d_in[9];
    const float* b_fc_o = (const float*)d_in[10];

    float* hidden = (float*)d_ws;           // (1024, 64) f32 = 256 KB
    float* outp   = (float*)d_out;          // 8 floats (softmax)

    // 64 batch tiles x 2 directions; 2 waves per block (one per h-col group).
    lstm_scan_kernel<<<dim3(64, 2), dim3(64), 0, stream>>>(
        x, W_ih_f, W_hh_f, b_f, W_ih_b, W_hh_b, b_b, hidden);

    head_kernel<<<dim3(1), dim3(1024), 0, stream>>>(
        hidden, W_fc_h, b_fc_h, W_fc_o, b_fc_o, outp);
}
